// Attention_65704409694537
// MI455X (gfx1250) — matrix operand order repaired and checked by reading, not compiled, not run on hardware
//
#include <hip/hip_runtime.h>

typedef _Float16 half4v  __attribute__((ext_vector_type(4)));
typedef _Float16 half8v  __attribute__((ext_vector_type(8)));
typedef _Float16 half16v __attribute__((ext_vector_type(16)));
typedef float    float8v __attribute__((ext_vector_type(8)));
typedef int      int4v   __attribute__((ext_vector_type(4)));

#define LDS_AS __attribute__((address_space(3)))
#define GLB_AS __attribute__((address_space(1)))

typedef GLB_AS int4v* g128_t;   // global pointer to 16B packet
typedef LDS_AS int4v* l128_t;   // LDS pointer to 16B packet

#if __has_builtin(__builtin_amdgcn_global_load_async_to_lds_b128) && \
    __has_builtin(__builtin_amdgcn_s_wait_asynccnt)
#define HAVE_ASYNC_LDS 1
#else
#define HAVE_ASYNC_LDS 0
#endif

__device__ __forceinline__ half16v cat8(half8v a, half8v b) {
    return __builtin_shufflevector(a, b, 0,1,2,3,4,5,6,7,8,9,10,11,12,13,14,15);
}

// ---------------------------------------------------------------------------
// f32 -> f16 conversion, 4 elements per thread (all sizes are multiples of 4)
// ---------------------------------------------------------------------------
__global__ __launch_bounds__(256)
void cvt_f32_to_f16_x4(const float* __restrict__ in, _Float16* __restrict__ out,
                       long n4) {
    long i = (long)blockIdx.x * blockDim.x + threadIdx.x;
    if (i >= n4) return;
    float4 v = ((const float4*)in)[i];
    half4v o;
    o.x = (_Float16)v.x; o.y = (_Float16)v.y;
    o.z = (_Float16)v.z; o.w = (_Float16)v.w;
    ((half4v*)out)[i] = o;
}

// ---------------------------------------------------------------------------
// D[M,N] = A[M,K] @ W[N,K]^T + bias[N]     (N, K compile-time)
// A, W f16 row-major (K contiguous). fp32 accumulation via WMMA.
// Block tile 128x128, BK=32 (one v_wmma_f32_16x16x32_f16 K-step),
// 8 waves as 4(M) x 2(N); per-wave 32x64 = 2x4 fragments.
// Staging: GLOBAL_LOAD_ASYNC_TO_LDS_B128 (ASYNCcnt) when available.
// ---------------------------------------------------------------------------
template <typename OutT, int N, int K>
__global__ __launch_bounds__(256, 2)
void wmma_gemm_nt_bias(const _Float16* __restrict__ A,
                       const _Float16* __restrict__ W,
                       const float*    __restrict__ bias,
                       OutT*           __restrict__ D) {
    constexpr int BM = 128, BN = 128, BK = 32, LDK = 40; // 80B rows, 16B aligned
    constexpr int NSTEP = K / BK;
    constexpr int ABUF  = BM * LDK;                       // elements per buffer
    __shared__ _Float16 As[2][ABUF];
    __shared__ _Float16 Ws[2][ABUF];

    const int tid  = threadIdx.x;
    const int lane = tid & 31;
    const int l15  = lane & 15;
    const int lhi  = lane >> 4;           // 0/1 K-half select (ISA fragment layout)
    const int wave = tid >> 5;            // 0..7
    const int wm   = (wave >> 1) * 32;    // wave M offset in block
    const int wn   = (wave & 1) * 64;     // wave N offset in block

    const int blockN = blockIdx.x * BN;
    const int blockM = blockIdx.y * BM;

    // Staging ownership: row = tid>>1 (0..127), 16-element half = tid&1.
    const int srow  = tid >> 1;
    const int shalf = (tid & 1) * 16;

    const _Float16* gA = A + (size_t)(blockM + srow) * K + shalf;
    const _Float16* gW = W + (size_t)(blockN + srow) * K + shalf;
    _Float16* sA0 = &As[0][srow * LDK + shalf];
    _Float16* sW0 = &Ws[0][srow * LDK + shalf];

    float8v acc[2][4] = {};

#if HAVE_ASYNC_LDS
    auto stage_async = [&](int buf, int k0) {
        _Float16* sa = sA0 + buf * ABUF;
        _Float16* sw = sW0 + buf * ABUF;
        __builtin_amdgcn_global_load_async_to_lds_b128(
            (g128_t)(gA + k0),     (l128_t)sa,       0, 0);
        __builtin_amdgcn_global_load_async_to_lds_b128(
            (g128_t)(gA + k0 + 8), (l128_t)(sa + 8), 0, 0);
        __builtin_amdgcn_global_load_async_to_lds_b128(
            (g128_t)(gW + k0),     (l128_t)sw,       0, 0);
        __builtin_amdgcn_global_load_async_to_lds_b128(
            (g128_t)(gW + k0 + 8), (l128_t)(sw + 8), 0, 0);
    };
#else
    half8v ra0, ra1, rw0, rw1;
    auto load_tile = [&](int k0) {
        ra0 = *(const half8v*)(gA + k0);  ra1 = *(const half8v*)(gA + k0 + 8);
        rw0 = *(const half8v*)(gW + k0);  rw1 = *(const half8v*)(gW + k0 + 8);
    };
    auto store_tile = [&](int buf) {
        _Float16* sa = sA0 + buf * ABUF;
        _Float16* sw = sW0 + buf * ABUF;
        *(half8v*)sa = ra0;  *(half8v*)(sa + 8) = ra1;
        *(half8v*)sw = rw0;  *(half8v*)(sw + 8) = rw1;
    };
#endif

    auto compute = [&](int cur) {
        // A fragment 16x32: lane l15 = row, lhi K-half.
        // v0..3 = K[lhi*8..+8), v4..7 = K[16+lhi*8..+8) -> two b128 LDS reads.
        half16v af[2];
        #pragma unroll
        for (int mb = 0; mb < 2; ++mb) {
            const _Float16* p = &As[cur][(wm + mb * 16 + l15) * LDK + lhi * 8];
            af[mb] = cat8(*(const half8v*)p, *(const half8v*)(p + 16));
        }
        // B fragment 32x16 (B[k,n] = W[n,k]): lane l15 = n, lhi K-group.
        half16v wf[4];
        #pragma unroll
        for (int nf = 0; nf < 4; ++nf) {
            const _Float16* p = &Ws[cur][(wn + nf * 16 + l15) * LDK + lhi * 8];
            wf[nf] = cat8(*(const half8v*)p, *(const half8v*)(p + 16));
        }
        #pragma unroll
        for (int mb = 0; mb < 2; ++mb)
            #pragma unroll
            for (int nf = 0; nf < 4; ++nf)
                acc[mb][nf] = __builtin_amdgcn_wmma_f32_16x16x32_f16(
                    false, af[mb], false, wf[nf], (short)0, acc[mb][nf],
                    false, false);
    };

#if HAVE_ASYNC_LDS
    stage_async(0, 0);
    __builtin_amdgcn_s_wait_asynccnt(0);
    __syncthreads();
    for (int step = 0; step < NSTEP; ++step) {
        const int cur = step & 1;
        if (step + 1 < NSTEP) {
            stage_async(cur ^ 1, (step + 1) * BK);   // other buffer: safe post-barrier
            if (step + 2 < NSTEP) {
                __builtin_prefetch(gA + (step + 2) * BK, 0, 3); // global_prefetch_b8
                __builtin_prefetch(gW + (step + 2) * BK, 0, 3);
            }
        }
        compute(cur);
        if (step + 1 < NSTEP) {
            __builtin_amdgcn_s_wait_asynccnt(0);
            __syncthreads();
        }
    }
#else
    load_tile(0);
    store_tile(0);
    __syncthreads();
    for (int step = 0; step < NSTEP; ++step) {
        const int cur = step & 1;
        if (step + 1 < NSTEP) {
            load_tile((step + 1) * BK);
            if (step + 2 < NSTEP) {
                __builtin_prefetch(gA + (step + 2) * BK, 0, 3);
                __builtin_prefetch(gW + (step + 2) * BK, 0, 3);
            }
        }
        compute(cur);
        __syncthreads();
        if (step + 1 < NSTEP) {
            store_tile(cur ^ 1);
            __syncthreads();
        }
    }
#endif

    // Epilogue: lane l15 = col, VGPR v -> row v + 8*lhi. N constexpr => imm offsets.
    #pragma unroll
    for (int nf = 0; nf < 4; ++nf) {
        const int col = blockN + wn + nf * 16 + l15;
        const float bv = bias[col];
        #pragma unroll
        for (int mb = 0; mb < 2; ++mb) {
            OutT* dp = D + (size_t)(blockM + wm + mb * 16 + 8 * lhi) * N + col;
            #pragma unroll
            for (int v = 0; v < 8; ++v)
                dp[(size_t)v * N] = (OutT)(acc[mb][nf][v] + bv);
        }
    }
}

// ---------------------------------------------------------------------------
// Attention core: one wave32 per (b, h). N=8 frames, hd=64.
// lane = n*4 + quad: n = query frame, quad = 16-wide head_dim slice.
// Scores reduced across quad lanes via shfl_xor; softmax in registers.
// qkv: [T, 2304] f16 (q:+0, k:+768, v:+1536, head offset h*64)
// ---------------------------------------------------------------------------
__global__ __launch_bounds__(128)
void attn_softmax_kernel(const _Float16* __restrict__ qkv,
                         const float*    __restrict__ relb,   // [15,12]
                         _Float16*       __restrict__ y,
                         int npairs) {
    const int wid = blockIdx.x * 4 + (threadIdx.x >> 5);
    if (wid >= npairs) return;
    const int lane = threadIdx.x & 31;
    const int b = wid / 12;
    const int h = wid - b * 12;
    const int n = lane >> 2;
    const int quad = lane & 3;
    const size_t tokQ = (size_t)(b * 8 + n);
    const int dcol = h * 64 + quad * 16;

    float qf[16];
    {
        const _Float16* qp = qkv + tokQ * 2304 + dcol;
        half8v q0 = *(const half8v*)qp, q1 = *(const half8v*)(qp + 8);
        #pragma unroll
        for (int i = 0; i < 8; ++i) { qf[i] = (float)q0[i]; qf[8 + i] = (float)q1[i]; }
    }

    float s[8];
    #pragma unroll
    for (int m = 0; m < 8; ++m) {
        const _Float16* kp = qkv + (size_t)(b * 8 + m) * 2304 + 768 + dcol;
        half8v k0 = *(const half8v*)kp, k1 = *(const half8v*)(kp + 8);
        float a = 0.f;
        #pragma unroll
        for (int i = 0; i < 8; ++i) { a += qf[i] * (float)k0[i]; a += qf[8 + i] * (float)k1[i]; }
        s[m] = a;
    }
    #pragma unroll
    for (int m = 0; m < 8; ++m) {
        s[m] += __shfl_xor(s[m], 1, 32);
        s[m] += __shfl_xor(s[m], 2, 32);
    }

    const float scale = 0.125f;   // 64^-0.5
    float mx = -1e30f;
    #pragma unroll
    for (int m = 0; m < 8; ++m) {
        s[m] = s[m] * scale + relb[(m - n + 7) * 12 + h];
        mx = fmaxf(mx, s[m]);
    }
    float sum = 0.f;
    #pragma unroll
    for (int m = 0; m < 8; ++m) { s[m] = __expf(s[m] - mx); sum += s[m]; }
    const float inv = 1.f / sum;

    float o[16] = {};
    #pragma unroll
    for (int m = 0; m < 8; ++m) {
        const _Float16* vp = qkv + (size_t)(b * 8 + m) * 2304 + 1536 + dcol;
        half8v v0 = *(const half8v*)vp, v1 = *(const half8v*)(vp + 8);
        const float p = s[m] * inv;
        #pragma unroll
        for (int i = 0; i < 8; ++i) { o[i] += p * (float)v0[i]; o[8 + i] += p * (float)v1[i]; }
    }

    _Float16* yp = y + tokQ * 768 + dcol;
    half8v o0, o1;
    #pragma unroll
    for (int i = 0; i < 8; ++i) { o0[i] = (_Float16)o[i]; o1[i] = (_Float16)o[8 + i]; }
    *(half8v*)yp = o0;
    *(half8v*)(yp + 8) = o1;
}

// ---------------------------------------------------------------------------
extern "C" void kernel_launch(void* const* d_in, const int* in_sizes, int n_in,
                              void* d_out, int out_size, void* d_ws, size_t ws_size,
                              hipStream_t stream) {
    const float* x      = (const float*)d_in[0];
    const float* qkv_w  = (const float*)d_in[1];
    const float* qkv_b  = (const float*)d_in[2];
    const float* proj_w = (const float*)d_in[3];
    const float* proj_b = (const float*)d_in[4];
    const float* relb   = (const float*)d_in[5];
    float* out = (float*)d_out;

    const int T  = 6272 * 8;   // 50176 tokens (multiple of 128)
    const int C  = 768;
    const int C3 = 2304;

    // Workspace (f16 intermediates; y reuses the xh region after x is dead):
    char* base = (char*)d_ws;
    _Float16* xh   = (_Float16*)(base);                                // T*C   (77,070,336 B)
    _Float16* wqh  = (_Float16*)(base + 77070336);                     // C3*C  ( 3,538,944 B)
    _Float16* wph  = (_Float16*)(base + 77070336 + 3538944);           // C*C   ( 1,179,648 B)
    _Float16* qkvh = (_Float16*)(base + 77070336 + 3538944 + 1179648); // T*C3  (231,211,008 B)
    _Float16* yh   = xh;

    {   // f32 -> f16 conversions
        long n4 = (long)T * C / 4;
        cvt_f32_to_f16_x4<<<(unsigned)((n4 + 255) / 256), 256, 0, stream>>>(x, xh, n4);
        n4 = (long)C3 * C / 4;
        cvt_f32_to_f16_x4<<<(unsigned)((n4 + 255) / 256), 256, 0, stream>>>(qkv_w, wqh, n4);
        n4 = (long)C * C / 4;
        cvt_f32_to_f16_x4<<<(unsigned)((n4 + 255) / 256), 256, 0, stream>>>(proj_w, wph, n4);
    }

    // qkv = x @ qkv_w^T + qkv_b  -> f16 [T, 2304]
    wmma_gemm_nt_bias<_Float16, 2304, 768><<<dim3(C3 / 128, T / 128), 256, 0, stream>>>(
        xh, wqh, qkv_b, qkvh);

    // attention + softmax + temporal relative bias -> y f16 [T, 768]
    {
        const int npairs = (T / 8) * 12;   // 75,264 (b,h) waves
        attn_softmax_kernel<<<npairs / 4, 128, 0, stream>>>(qkvh, relb, yh, npairs);
    }

    // out = y @ proj_w^T + proj_b -> f32 [T, 768]
    wmma_gemm_nt_bias<float, 768, 768><<<dim3(C / 128, T / 128), 256, 0, stream>>>(
        yh, wph, proj_b, out);
}
